// LinearNemotronHMoE_10419590660255
// MI455X (gfx1250) — matrix operand order repaired and checked by reading, not compiled, not run on hardware
//
#include <hip/hip_runtime.h>
#include <stdint.h>

// ---------------------------------------------------------------------------
// Problem constants (match reference config)
// ---------------------------------------------------------------------------
constexpr int Dh   = 2048;   // hidden size
constexpr int Hm   = 1024;   // moe intermediate
constexpr int HSm  = 4096;   // shared intermediate
constexpr int Ne   = 32;     // experts
constexpr int Gg   = 8;      // groups
constexpr int CapE = 1024;   // capacity
constexpr int Tt   = 4096;   // tokens (B*S)
constexpr int KTOP = 4;

#define AS1 __attribute__((address_space(1)))
#define AS3 __attribute__((address_space(3)))

#if __has_builtin(__builtin_amdgcn_global_load_async_to_lds_b128)
#define ASYNC_LDS 1
#else
#define ASYNC_LDS 0
#endif

__device__ __forceinline__ void wait_asynccnt0() {
#if __has_builtin(__builtin_amdgcn_s_wait_asynccnt)
  __builtin_amdgcn_s_wait_asynccnt(0);
#else
  asm volatile("s_wait_asynccnt 0x0" ::: "memory");
#endif
}

// ---------------------------------------------------------------------------
// Trivial vector types (union-safe, guaranteed vector codegen)
// ---------------------------------------------------------------------------
typedef __attribute__((ext_vector_type(16))) __bf16 v16bf;
typedef __attribute__((ext_vector_type(8)))  float  v8f;
typedef __attribute__((ext_vector_type(4)))  float  v4f;
typedef __attribute__((ext_vector_type(4)))  unsigned int v4u;
typedef __attribute__((ext_vector_type(4)))  int v4i;

union FragU { v4u q[2]; v16bf v; };

__device__ __forceinline__ unsigned int bf1(float x) {
  return (__float_as_uint(x) + 0x8000u) >> 16;  // round-half-up fp32->bf16
}
// pack two fp32 into packed bf16x2 with one v_perm_b32
__device__ __forceinline__ unsigned int bfpair(float a, float b) {
  unsigned int ua = __float_as_uint(a) + 0x8000u;
  unsigned int ub = __float_as_uint(b) + 0x8000u;
  // result = { ub[31:16], ua[31:16] } : bytes 2,3 of S1(=ua), bytes 2,3 of S0(=ub)
  return __builtin_amdgcn_perm(ub, ua, 0x07060302u);
}

// ---------------------------------------------------------------------------
// Gate GEMM + grouped top-k routing, one 128-thread block per token
// ---------------------------------------------------------------------------
__global__ __launch_bounds__(128) void gate_route_kernel(
    const float* __restrict__ x, const float* __restrict__ gw,
    const float* __restrict__ bias, int* __restrict__ tidx,
    float* __restrict__ tw) {
  __shared__ float red[128];
  __shared__ float sc[Ne];
  __shared__ float scc[Ne];
  const int t = blockIdx.x;
  const int tid = threadIdx.x;
  const int e = tid & 31;
  const int ch = tid >> 5;  // 4 chunks of 512
  const v4f* xv = (const v4f*)(x + (size_t)t * Dh + ch * 512);
  const v4f* gv = (const v4f*)(gw + (size_t)e * Dh + ch * 512);
  float s = 0.f;
#pragma unroll 8
  for (int i = 0; i < 128; ++i) {
    v4f a = xv[i], b = gv[i];
    s += a.x * b.x + a.y * b.y + a.z * b.z + a.w * b.w;
  }
  red[tid] = s;
  __syncthreads();
  if (tid < Ne) {
    float v = red[tid] + red[tid + 32] + red[tid + 64] + red[tid + 96];
    float sig = 1.f / (1.f + expf(-v));
    sc[tid] = sig;
    scc[tid] = sig + bias[tid];
  }
  __syncthreads();
  if (tid == 0) {
    // group score = sum of top-2 within each group of 4
    float gs[Gg];
#pragma unroll
    for (int g = 0; g < Gg; ++g) {
      float m1 = -1e30f, m2 = -1e30f;
#pragma unroll
      for (int j = 0; j < Ne / Gg; ++j) {
        float v = scc[g * (Ne / Gg) + j];
        if (v > m1) { m2 = m1; m1 = v; } else if (v > m2) { m2 = v; }
      }
      gs[g] = m1 + m2;
    }
    bool gsel[Gg] = {false, false, false, false, false, false, false, false};
#pragma unroll
    for (int it = 0; it < 4; ++it) {  // TKG = 4 groups
      float best = -1e30f; int bi = 0;
      for (int g = 0; g < Gg; ++g)
        if (!gsel[g] && gs[g] > best) { best = gs[g]; bi = g; }
      gsel[bi] = true;
    }
    float ms[Ne];
    for (int i = 0; i < Ne; ++i) ms[i] = gsel[i >> 2] ? scc[i] : 0.0f;
    int idxv[KTOP]; float wv[KTOP]; float wsum = 0.f;
#pragma unroll
    for (int it = 0; it < KTOP; ++it) {
      float best = -1.f; int bi = 0;
      for (int i = 0; i < Ne; ++i)
        if (ms[i] > best) { best = ms[i]; bi = i; }
      ms[bi] = -2.f;
      idxv[it] = bi;
      wv[it] = sc[bi];          // weight from raw sigmoid score
      wsum += sc[bi];
    }
    float inv = 2.5f / (wsum + 1e-20f);  // norm + routed_scaling_factor
#pragma unroll
    for (int it = 0; it < KTOP; ++it) {
      tidx[t * KTOP + it] = idxv[it];
      tw[t * KTOP + it] = wv[it] * inv;
    }
  }
}

// ---------------------------------------------------------------------------
// Dispatch: atomic slot assignment (order-free; equivalent under no overflow)
// ---------------------------------------------------------------------------
__global__ __launch_bounds__(256) void dispatch_kernel(
    const int* __restrict__ tidx, const float* __restrict__ tw,
    int* __restrict__ counts, int* __restrict__ etok, float* __restrict__ ew) {
  int t = blockIdx.x * 256 + threadIdx.x;
  if (t >= Tt) return;
#pragma unroll
  for (int k = 0; k < KTOP; ++k) {
    int e = tidx[t * KTOP + k];
    float w = tw[t * KTOP + k];
    int pos = atomicAdd(&counts[e], 1);
    if (pos < CapE) {
      etok[e * CapE + pos] = t;
      ew[e * CapE + pos] = w;
    }
  }
}

// ---------------------------------------------------------------------------
// Generic WMMA bf16 GEMM: C[M,N] = A[M,K] * W[N,K]^T, 128x128x32 block tile,
// double-buffered LDS, 8 waves (2x4), wave tile 64x32 = 8 accumulators of
// v_wmma_f32_16x16x32_bf16.
// MODE 0: shared-up   A=flat f32,      out = relu2 -> sact bf16   (K=Dh)
// MODE 1: shared-dn   A=sact bf16,     out = f32 store -> out     (K=HSm)
// MODE 2: expert-up   A=gathered f32,  out = relu2 -> hact bf16   (K=Dh)
// MODE 3: expert-dn   A=hact bf16,     out = atomicAdd(w * v)     (K=Hm)
// bf16 A tiles (modes 1/3) are staged with global_load_async_to_lds_b128
// (ASYNCcnt) when the toolchain exposes the builtin.
// ---------------------------------------------------------------------------
template <int MODE>
__global__ __launch_bounds__(256) void gemm_wmma(
    const float* __restrict__ x, const float* __restrict__ w_sup,
    const float* __restrict__ w_sdn, const float* __restrict__ w_up,
    const float* __restrict__ w_dn, unsigned short* __restrict__ hact,
    unsigned short* __restrict__ sact, float* __restrict__ out,
    const int* __restrict__ etok, const float* __restrict__ ew,
    const int* __restrict__ counts) {
  constexpr int BM = 128, BN = 128, BK = 32, LDT = 40;  // padded LDS row
  constexpr int Kd = (MODE == 0 || MODE == 2) ? Dh : (MODE == 1 ? HSm : Hm);
  constexpr bool A_BF16 = (MODE == 1 || MODE == 3);
  constexpr bool USE_ASYNC = ASYNC_LDS && A_BF16;

  __shared__ __attribute__((aligned(16))) unsigned short As[2][BM * LDT];
  __shared__ __attribute__((aligned(16))) unsigned short Bs[2][BN * LDT];
  __shared__ int   ttile[BM];
  __shared__ float wtile[BM];

  const int tid = threadIdx.x;
  const int lane = tid & 31;
  const int wid = tid >> 5;
  const int n0 = blockIdx.x * BN;
  const int m0 = blockIdx.y * BM;
  const int e = blockIdx.z;

  int cnt = 0;
  if constexpr (MODE == 2 || MODE == 3) {
    cnt = counts[e];
    if (cnt > CapE) cnt = CapE;
    if (m0 >= cnt) return;  // uniform early-exit for empty M tiles
    if (tid < BM) {
      int c = m0 + tid; int tk = 0; float wv = 0.f;
      if (c < cnt) { tk = etok[e * CapE + c]; wv = ew[e * CapE + c]; }
      ttile[tid] = tk;
      wtile[tid] = wv;
    }
    __syncthreads();
  }

  const float* wbase = (MODE == 0)   ? w_sup
                       : (MODE == 1) ? w_sdn
                       : (MODE == 2) ? (w_up + (size_t)e * Hm * Dh)
                                     : (w_dn + (size_t)e * Dh * Hm);

  // staging: each thread owns 2 (row, 8-element K unit) pairs per tile
  const int srow = tid >> 2;       // 0..63
  const int sku = (tid & 3) * 8;   // 0,8,16,24

  const float* aF[2] = {nullptr, nullptr};
  const unsigned short* aH[2] = {nullptr, nullptr};
#pragma unroll
  for (int p = 0; p < 2; ++p) {
    int row = srow + p * 64;
    if constexpr (MODE == 0)
      aF[p] = x + (size_t)(m0 + row) * Dh;
    else if constexpr (MODE == 2)
      aF[p] = x + (size_t)ttile[row] * Dh;  // gather
    else if constexpr (MODE == 1)
      aH[p] = sact + (size_t)(m0 + row) * HSm;
    else
      aH[p] = hact + ((size_t)e * CapE + (m0 + row)) * Hm;
  }
  const float* bF[2];
#pragma unroll
  for (int p = 0; p < 2; ++p) bF[p] = wbase + (size_t)(n0 + srow + p * 64) * Kd;

  v4f fA[2][2]; v4u uA[2]; v4f fB[2][2];

  auto gloadB = [&](int k0) {
#pragma unroll
    for (int p = 0; p < 2; ++p) {
      const float* sb = bF[p] + k0 + sku;
      fB[p][0] = *(const v4f*)sb;
      fB[p][1] = *(const v4f*)(sb + 4);
    }
  };
  auto gloadA = [&](int k0) {  // register-path A fetch
#pragma unroll
    for (int p = 0; p < 2; ++p) {
      if constexpr (A_BF16) {
        uA[p] = *(const v4u*)(aH[p] + k0 + sku);
      } else {
        const float* s = aF[p] + k0 + sku;
        fA[p][0] = *(const v4f*)s;
        fA[p][1] = *(const v4f*)(s + 4);
      }
    }
  };
#if ASYNC_LDS
  auto asyncA = [&](int k0, int buf) {  // bf16 A: global -> LDS, no VGPRs
#pragma unroll
    for (int p = 0; p < 2; ++p) {
      int row = srow + p * 64;
      const unsigned short* g = aH[p] + k0 + sku;
      unsigned short* l = &As[buf][row * LDT + sku];
      __builtin_amdgcn_global_load_async_to_lds_b128(
          (AS1 v4i*)g, (AS3 v4i*)l, 0, 0);
    }
  };
#endif
  auto lstoreB = [&](int buf) {
#pragma unroll
    for (int p = 0; p < 2; ++p) {
      int row = srow + p * 64;
      v4u vb;
      vb.x = bfpair(fB[p][0].x, fB[p][0].y);
      vb.y = bfpair(fB[p][0].z, fB[p][0].w);
      vb.z = bfpair(fB[p][1].x, fB[p][1].y);
      vb.w = bfpair(fB[p][1].z, fB[p][1].w);
      *(v4u*)&Bs[buf][row * LDT + sku] = vb;
    }
  };
  auto lstoreA = [&](int buf) {
#pragma unroll
    for (int p = 0; p < 2; ++p) {
      int row = srow + p * 64;
      v4u va;
      if constexpr (A_BF16) {
        va = uA[p];
      } else {
        va.x = bfpair(fA[p][0].x, fA[p][0].y);
        va.y = bfpair(fA[p][0].z, fA[p][0].w);
        va.z = bfpair(fA[p][1].x, fA[p][1].y);
        va.w = bfpair(fA[p][1].z, fA[p][1].w);
      }
      *(v4u*)&As[buf][row * LDT + sku] = va;
    }
  };

  const int rbase = (wid >> 2) * 64;  // wave M origin: 0 or 64
  const int cbase = (wid & 3) * 32;   // wave N origin: 0,32,64,96

  v8f acc[4][2];
  const v8f vzero = {0.f, 0.f, 0.f, 0.f, 0.f, 0.f, 0.f, 0.f};
#pragma unroll
  for (int mi = 0; mi < 4; ++mi)
#pragma unroll
    for (int ni = 0; ni < 2; ++ni) acc[mi][ni] = vzero;

  // ---- prologue: fill buffer 0 ----
  gloadB(0);
#if ASYNC_LDS
  if constexpr (USE_ASYNC) asyncA(0, 0); else gloadA(0);
#else
  gloadA(0);
#endif
  lstoreB(0);
  if constexpr (!USE_ASYNC) lstoreA(0);
  if constexpr (USE_ASYNC) wait_asynccnt0();
  __syncthreads();

  int cur = 0;
  for (int k0 = 0; k0 < Kd; k0 += BK) {
    const int nxt = cur ^ 1;
    const bool more = (k0 + BK) < Kd;
    if (more) {  // kick off next tile before touching this one
#if ASYNC_LDS
      if constexpr (USE_ASYNC) asyncA(k0 + BK, nxt); else gloadA(k0 + BK);
#else
      gloadA(k0 + BK);
#endif
      gloadB(k0 + BK);
      __builtin_prefetch((const void*)(bF[0] + k0 + 2 * BK), 0, 1);
      __builtin_prefetch((const void*)(bF[1] + k0 + 2 * BK), 0, 1);
    }

    // fragment loads per CDNA5 16-bit A(16x32)/B(32x16) lane layouts
    const unsigned short* Ac = As[cur];
    const unsigned short* Bc = Bs[cur];
    const int lrow = lane & 15;
    const int half = lane >> 4;
    FragU a[4], b[2];
#pragma unroll
    for (int mi = 0; mi < 4; ++mi) {
      const unsigned short* p = &Ac[(rbase + mi * 16 + lrow) * LDT + half * 8];
      a[mi].q[0] = *(const v4u*)p;         // K half*8 .. half*8+7
      a[mi].q[1] = *(const v4u*)(p + 16);  // K 16+half*8 ..
    }
#pragma unroll
    for (int ni = 0; ni < 2; ++ni) {
      const unsigned short* p = &Bc[(cbase + ni * 16 + lrow) * LDT + half * 8];
      b[ni].q[0] = *(const v4u*)p;        // column lrow, K half*8 .. half*8+7
      b[ni].q[1] = *(const v4u*)(p + 16);
    }
#pragma unroll
    for (int mi = 0; mi < 4; ++mi)
#pragma unroll
      for (int ni = 0; ni < 2; ++ni)
        acc[mi][ni] = __builtin_amdgcn_wmma_f32_16x16x32_bf16(
            false, a[mi].v, false, b[ni].v, (short)0, acc[mi][ni], false,
            false);

    if (more) {  // fill the other buffer while WMMAs drain
      lstoreB(nxt);
      if constexpr (!USE_ASYNC) lstoreA(nxt);
      if constexpr (USE_ASYNC) wait_asynccnt0();
    }
    __syncthreads();
    cur = nxt;
  }

  // Epilogue. C layout: vgpr r, lanes 0-15 -> M=r, lanes 16-31 -> M=8+r; N=lane&15
#pragma unroll
  for (int mi = 0; mi < 4; ++mi) {
#pragma unroll
    for (int ni = 0; ni < 2; ++ni) {
      const int lr0 = rbase + mi * 16 + ((lane & 16) ? 8 : 0);
      const int gc = n0 + cbase + ni * 16 + (lane & 15);
#pragma unroll
      for (int r = 0; r < 8; ++r) {
        const int lrl = lr0 + r;   // row within block tile
        const int gm = m0 + lrl;   // global row
        float v = acc[mi][ni][r];
        if constexpr (MODE == 0) {
          float rv = fmaxf(v, 0.f); rv *= rv;
          sact[(size_t)gm * HSm + gc] = (unsigned short)bf1(rv);
        } else if constexpr (MODE == 1) {
          out[(size_t)gm * Dh + gc] = v;
        } else if constexpr (MODE == 2) {
          float rv = fmaxf(v, 0.f); rv *= rv;
          hact[((size_t)e * CapE + gm) * Hm + gc] = (unsigned short)bf1(rv);
        } else {
          if (gm < cnt) {
            int tkn = ttile[lrl];
            float w = wtile[lrl];
            atomicAdd(out + (size_t)tkn * Dh + gc, v * w);
          }
        }
      }
    }
  }
}

// ---------------------------------------------------------------------------
// Host-side launch (graph-capture safe: only async ops on `stream`)
// ---------------------------------------------------------------------------
extern "C" void kernel_launch(void* const* d_in, const int* in_sizes, int n_in,
                              void* d_out, int out_size, void* d_ws,
                              size_t ws_size, hipStream_t stream) {
  (void)in_sizes; (void)n_in; (void)out_size; (void)ws_size;
  const float* x   = (const float*)d_in[0];  // [2,2048,2048]
  const float* gw  = (const float*)d_in[1];  // [32,2048]
  const float* bc  = (const float*)d_in[2];  // [32]
  const float* upw = (const float*)d_in[3];  // [32,1024,2048]
  const float* dnw = (const float*)d_in[4];  // [32,2048,1024]
  const float* suw = (const float*)d_in[5];  // [4096,2048]
  const float* sdw = (const float*)d_in[6];  // [2048,4096]
  float* out = (float*)d_out;

  char* ws = (char*)d_ws;
  size_t off = 0;
  int* counts = (int*)(ws + off); off += 256;
  int* tidx   = (int*)(ws + off); off += (size_t)Tt * KTOP * 4;
  float* tw   = (float*)(ws + off); off += (size_t)Tt * KTOP * 4;
  int* etok   = (int*)(ws + off); off += (size_t)Ne * CapE * 4;
  float* ew   = (float*)(ws + off); off += (size_t)Ne * CapE * 4;
  unsigned short* hact = (unsigned short*)(ws + off);
  off += (size_t)Ne * CapE * Hm * 2;                      // 64 MB
  unsigned short* sact = (unsigned short*)(ws + off);
  off += (size_t)Tt * HSm * 2;                            // 32 MB

  (void)hipMemsetAsync(counts, 0, 256, stream);
  gate_route_kernel<<<Tt, 128, 0, stream>>>(x, gw, bc, tidx, tw);
  dispatch_kernel<<<Tt / 256, 256, 0, stream>>>(tidx, tw, counts, etok, ew);

  // shared expert: up (relu2 -> sact), then down writes full `out`
  gemm_wmma<0><<<dim3(HSm / 128, Tt / 128, 1), 256, 0, stream>>>(
      x, suw, sdw, upw, dnw, hact, sact, out, etok, ew, counts);
  gemm_wmma<1><<<dim3(Dh / 128, Tt / 128, 1), 256, 0, stream>>>(
      x, suw, sdw, upw, dnw, hact, sact, out, etok, ew, counts);

  // routed experts: up (gather, relu2 -> hact), down (scatter-add into out)
  gemm_wmma<2><<<dim3(Hm / 128, CapE / 128, Ne), 256, 0, stream>>>(
      x, suw, sdw, upw, dnw, hact, sact, out, etok, ew, counts);
  gemm_wmma<3><<<dim3(Dh / 128, CapE / 128, Ne), 256, 0, stream>>>(
      x, suw, sdw, upw, dnw, hact, sact, out, etok, ew, counts);
}
